// TMix_22179211116820
// MI455X (gfx1250) — matrix operand order repaired and checked by reading, not compiled, not run on hardware
//
#include <hip/hip_runtime.h>
#include <hip/hip_bf16.h>
#include <cstdint>

// Problem constants (match reference)
#define BB 4
#define TT 1024
#define CC 1024
#define HH 16
#define DD 64
#define MM (BB * TT)                     // 4096 rows
#define LOG_DECAY_SCALE (-0.6065306597126334f)
#define GN_EPS (64e-5f)

typedef __attribute__((ext_vector_type(16))) __bf16 v16bf;
typedef __attribute__((ext_vector_type(8)))  float  v8f;

#if defined(__has_builtin)
# if __has_builtin(__builtin_amdgcn_global_load_async_to_lds_b32)
#  define HAVE_ASYNC_LDS 1
# endif
#endif

#ifdef HAVE_ASYNC_LDS
typedef __attribute__((address_space(1))) int* gas_int_p;   // global
typedef __attribute__((address_space(3))) int* las_int_p;   // LDS
#endif

// ---------------------------------------------------------------------------
// helpers
// ---------------------------------------------------------------------------
__device__ __forceinline__ unsigned short f2bf(float x) {
  unsigned int u = __float_as_uint(x);
  u += 0x7FFFu + ((u >> 16) & 1u);       // round-to-nearest-even
  return (unsigned short)(u >> 16);
}

union FragBF { uint4 q[2]; v16bf v; };

// A fragment: 16x32 bf16 (MxK). ISA layout: lanes 0-15 hold row M=lane,
// K=0..7 in v[0..3] and K=16..23 in v[4..7]; lanes 16-31 get K+8.
__device__ __forceinline__ v16bf load_frag_a(const unsigned short* __restrict__ A,
                                             int ldk, int row0, int kb, int lane) {
  const unsigned short* p =
      A + (size_t)(row0 + (lane & 15)) * ldk + kb + ((lane >> 4) << 3);
  FragBF f;
  f.q[0] = *(const uint4*)(p);
  f.q[1] = *(const uint4*)(p + 16);
  return f.v;
}

// B fragment: 32x16 bf16 (KxN), column per lane, same K layout as A (K=0..7,16..23; lanes 16-31 get K+8).
// W stored [N,K] row-major -> column n of B is row n of W.
__device__ __forceinline__ v16bf load_frag_b(const unsigned short* __restrict__ W,
                                             int ldk, int col0, int kb, int lane) {
  const unsigned short* p =
      W + (size_t)(col0 + (lane & 15)) * ldk + kb + ((lane >> 4) << 3);
  FragBF f;
  f.q[0] = *(const uint4*)(p);
  f.q[1] = *(const uint4*)(p + 16);
  return f.v;
}

// ---------------------------------------------------------------------------
// WMMA GEMM: OUT[M,N] = A[M,K] @ W[N,K]^T  (bf16 in, f32 accumulate)
// One wave per block; wave tile (16*MT)x(16*NT) of WMMA 16x16x32 tiles.
// mode: 0 = identity, 1 = tanh, 2 = scale*sigmoid(x + bias[col])
// out_bf16: 1 -> write bf16 (feeds a following GEMM), 0 -> write f32
// ---------------------------------------------------------------------------
template <int MT, int NT>
__global__ __launch_bounds__(32) void wmma_gemm_bf16(
    const unsigned short* __restrict__ A, const unsigned short* __restrict__ W,
    void* __restrict__ OUT, int M, int N, int K,
    int mode, const float* __restrict__ bias, float scale, int out_bf16) {
  const int lane = threadIdx.x;
  const int m0 = blockIdx.x * (16 * MT);
  const int n0 = blockIdx.y * (16 * NT);

  v8f acc[MT][NT];
#pragma unroll
  for (int i = 0; i < MT; i++)
#pragma unroll
    for (int j = 0; j < NT; j++) { v8f z = {}; acc[i][j] = z; }

  for (int kb = 0; kb < K; kb += 32) {
    if (kb + 128 < K) {   // uniform condition; pull next tiles into WGP caches
      __builtin_prefetch(A + (size_t)(m0 + (lane & 15)) * K + kb + 128, 0, 3);
      __builtin_prefetch(W + (size_t)(n0 + (lane & 15)) * K + kb + 128, 0, 3);
    }
    v16bf af[MT], bfr[NT];
#pragma unroll
    for (int i = 0; i < MT; i++) af[i] = load_frag_a(A, K, m0 + 16 * i, kb, lane);
#pragma unroll
    for (int j = 0; j < NT; j++) bfr[j] = load_frag_b(W, K, n0 + 16 * j, kb, lane);
#pragma unroll
    for (int i = 0; i < MT; i++)
#pragma unroll
      for (int j = 0; j < NT; j++)
        acc[i][j] = __builtin_amdgcn_wmma_f32_16x16x32_bf16(
            false, af[i], false, bfr[j], (short)0, acc[i][j], false, false);
  }

  const int rb = ((lane >> 4) << 3);  // lanes 16-31 hold rows +8
  const int cb = lane & 15;
  float* outf = (float*)OUT;
  unsigned short* outb = (unsigned short*)OUT;

#pragma unroll
  for (int i = 0; i < MT; i++) {
#pragma unroll
    for (int j = 0; j < NT; j++) {
      const int r0 = m0 + 16 * i + rb;
      const int c  = n0 + 16 * j + cb;
      const float bv = (mode == 2 && bias) ? bias[c] : 0.f;
#pragma unroll
      for (int e = 0; e < 8; e++) {
        float v = acc[i][j][e];
        if (mode == 1)      v = tanhf(v);
        else if (mode == 2) v = scale / (1.f + __expf(-(v + bv)));
        const size_t oidx = (size_t)(r0 + e) * N + c;
        if (out_bf16) outb[oidx] = f2bf(v);
        else          outf[oidx] = v;
      }
    }
  }
}

// ---------------------------------------------------------------------------
// Token shift + six mixes, fused; writes bf16 activations for the GEMMs.
// ---------------------------------------------------------------------------
__global__ void mix_kernel(const float* __restrict__ x,
                           const float* __restrict__ mr, const float* __restrict__ mw,
                           const float* __restrict__ mk, const float* __restrict__ mv,
                           const float* __restrict__ ma, const float* __restrict__ mg,
                           unsigned short* __restrict__ xr, unsigned short* __restrict__ xw,
                           unsigned short* __restrict__ xk, unsigned short* __restrict__ xv,
                           unsigned short* __restrict__ xa, unsigned short* __restrict__ xg) {
  const int idx = blockIdx.x * 256 + threadIdx.x;
  if (idx >= MM * CC) return;
  const int c = idx & (CC - 1);
  const int t = (idx >> 10) & (TT - 1);
  const float xv_ = x[idx];
  const float prev = (t > 0) ? x[idx - CC] : 0.f;
  const float xx = prev - xv_;
  xr[idx] = f2bf(xv_ + xx * mr[c]);
  xw[idx] = f2bf(xv_ + xx * mw[c]);
  xk[idx] = f2bf(xv_ + xx * mk[c]);
  xv[idx] = f2bf(xv_ + xx * mv[c]);
  xa[idx] = f2bf(xv_ + xx * ma[c]);
  xg[idx] = f2bf(xv_ + xx * mg[c]);
}

__global__ void cvt_bf16(const float* __restrict__ src, unsigned short* __restrict__ dst, int n) {
  const int i = blockIdx.x * 256 + threadIdx.x;
  if (i < n) dst[i] = f2bf(src[i]);
}

// src [rows,cols] f32 -> dst [cols,rows] bf16 (for [K,N]-stored low-rank weights)
__global__ void cvt_bf16_tr(const float* __restrict__ src, unsigned short* __restrict__ dst,
                            int rows, int cols) {
  const int i = blockIdx.x * 256 + threadIdx.x;
  if (i >= rows * cols) return;
  const int r = i / cols, c = i - r * cols;
  dst[(size_t)c * rows + r] = f2bf(src[i]);
}

// ---------------------------------------------------------------------------
// per-(b,t,h) prep: kk normalization -> -kk, kk*a ; k2 ; value mixing (in place)
// 256 threads = 4 head-rows of 64
// ---------------------------------------------------------------------------
__device__ __forceinline__ float grp_reduce64(float* red, int tid, int d, float v) {
  red[tid] = v;
  __syncthreads();
#pragma unroll
  for (int s = 32; s > 0; s >>= 1) {
    if (d < s) red[tid] += red[tid + s];
    __syncthreads();
  }
  const float r = red[tid & ~63];
  __syncthreads();
  return r;
}

__global__ __launch_bounds__(256) void prep_kernel(
    float* __restrict__ k_inout,            // in: k, out: k2
    const float* __restrict__ a,
    float* __restrict__ v_inout,            // in: v_raw, out: mixed v
    const float* __restrict__ v_first,
    const float* __restrict__ vmix_sig,     // sigmoid(v0 + (xv@v1)@v2)
    float* __restrict__ nkk, float* __restrict__ kkah,
    const float* __restrict__ k_k, const float* __restrict__ k_a) {
  __shared__ float red[256];
  const int tid = threadIdx.x;
  const int grp = blockIdx.x * 4 + (tid >> 6);   // (b*T+t)*H + h
  const int d = tid & 63;
  const size_t idx = (size_t)grp * DD + d;
  const int c = (int)(idx & (CC - 1));

  const float kv = k_inout[idx];
  const float av = a[idx];
  const float kkv = kv * k_k[c];

  const float ss = grp_reduce64(red, tid, d, kkv * kkv);
  const float inv = 1.f / fmaxf(sqrtf(ss), 1e-12f);

  nkk[idx]  = -kkv * inv;
  kkah[idx] =  kkv * inv * av;
  k_inout[idx] = kv * (1.f + (av - 1.f) * k_a[c]);

  const float vr = v_inout[idx];
  v_inout[idx] = vr + (v_first[idx] - vr) * vmix_sig[idx];
}

// ---------------------------------------------------------------------------
// Sequential RWKV7 scan. One block per (b,h); thread d owns state row S[d][*]
// kept fully in registers (64 f32). Per-step vectors broadcast via LDS,
// staged with GLOBAL_LOAD_ASYNC_TO_LDS when the toolchain exposes it.
// ---------------------------------------------------------------------------
__global__ __launch_bounds__(64) void scan_kernel(
    const float* __restrict__ r, const float* __restrict__ wlog,
    const float* __restrict__ k2, const float* __restrict__ v,
    const float* __restrict__ nkk, const float* __restrict__ kkah,
    float* __restrict__ o) {
  const int bh = blockIdx.x;
  const int b = bh >> 4, h = bh & 15;
  const int d = threadIdx.x;

  __shared__ float r_s[DD], ew_s[DD], k_s[DD], v_s[DD], a_s[DD], b_s[DD];

  float S[DD];
#pragma unroll
  for (int i = 0; i < DD; i++) S[i] = 0.f;

  const size_t base = ((size_t)b * TT) * CC + (size_t)h * DD;
  for (int t = 0; t < TT; t++) {
    const size_t p = base + (size_t)t * CC;
#ifdef HAVE_ASYNC_LDS
    // async DMA straight into LDS (ASYNCcnt-tracked, no VGPR staging)
    __builtin_amdgcn_global_load_async_to_lds_b32(
        (gas_int_p)(r + p + d), (las_int_p)(&r_s[d]), 0, 0);
    __builtin_amdgcn_global_load_async_to_lds_b32(
        (gas_int_p)(k2 + p + d), (las_int_p)(&k_s[d]), 0, 0);
    __builtin_amdgcn_global_load_async_to_lds_b32(
        (gas_int_p)(v + p + d), (las_int_p)(&v_s[d]), 0, 0);
    __builtin_amdgcn_global_load_async_to_lds_b32(
        (gas_int_p)(nkk + p + d), (las_int_p)(&a_s[d]), 0, 0);
    __builtin_amdgcn_global_load_async_to_lds_b32(
        (gas_int_p)(kkah + p + d), (las_int_p)(&b_s[d]), 0, 0);
    ew_s[d] = __expf(wlog[p + d]);   // needs VALU exp anyway
# if defined(__has_builtin) && __has_builtin(__builtin_amdgcn_s_wait_asynccnt)
    __builtin_amdgcn_s_wait_asynccnt(0);
# else
    asm volatile("s_wait_asynccnt 0x0" ::: "memory");
# endif
#else
    r_s[d]  = r[p + d];
    k_s[d]  = k2[p + d];
    v_s[d]  = v[p + d];
    a_s[d]  = nkk[p + d];
    b_s[d]  = kkah[p + d];
    ew_s[d] = __expf(wlog[p + d]);
#endif
    __syncthreads();

    float sab = 0.f;
#pragma unroll
    for (int kk = 0; kk < DD; kk++) sab += S[kk] * a_s[kk];

    const float vt = v_s[d];
    float ot = 0.f;
#pragma unroll
    for (int kk = 0; kk < DD; kk++) {
      const float s = S[kk] * ew_s[kk] + sab * b_s[kk] + vt * k_s[kk];
      S[kk] = s;
      ot += s * r_s[kk];
    }
    o[p + d] = ot;
    __syncthreads();
  }
}

// ---------------------------------------------------------------------------
// GroupNorm(per head) + bonus + gate, emits bf16 y for the final WMMA GEMM.
// ---------------------------------------------------------------------------
__global__ __launch_bounds__(256) void norm_bonus_gate(
    const float* __restrict__ o, const float* __restrict__ r,
    const float* __restrict__ k2, const float* __restrict__ v,
    const float* __restrict__ g, const float* __restrict__ r_k,
    const float* __restrict__ gn_w, const float* __restrict__ gn_b,
    unsigned short* __restrict__ y) {
  __shared__ float red[256];
  const int tid = threadIdx.x;
  const int grp = blockIdx.x * 4 + (tid >> 6);
  const int d = tid & 63;
  const size_t idx = (size_t)grp * DD + d;
  const int c = (int)(idx & (CC - 1));

  const float ov = o[idx];
  const float s1 = grp_reduce64(red, tid, d, ov);
  const float s2 = grp_reduce64(red, tid, d, ov * ov);
  const float dot = grp_reduce64(red, tid, d, r[idx] * k2[idx] * r_k[c]);

  const float mean = s1 * (1.f / DD);
  const float var = s2 * (1.f / DD) - mean * mean;
  float on = (ov - mean) * rsqrtf(var + GN_EPS);
  on = on * gn_w[c] + gn_b[c];
  const float bonus = dot * v[idx];
  y[idx] = f2bf((on + bonus) * g[idx]);
}

// ---------------------------------------------------------------------------
// host side
// ---------------------------------------------------------------------------
extern "C" void kernel_launch(void* const* d_in, const int* in_sizes, int n_in,
                              void* d_out, int out_size, void* d_ws, size_t ws_size,
                              hipStream_t stream) {
  (void)in_sizes; (void)n_in; (void)out_size; (void)ws_size;
  const float* x       = (const float*)d_in[0];
  const float* v_first = (const float*)d_in[1];
  const float* x_r = (const float*)d_in[2];
  const float* x_w = (const float*)d_in[3];
  const float* x_k = (const float*)d_in[4];
  const float* x_v = (const float*)d_in[5];
  const float* x_a = (const float*)d_in[6];
  const float* x_g = (const float*)d_in[7];
  const float* Wr = (const float*)d_in[8];
  const float* Wk = (const float*)d_in[9];
  const float* Wv = (const float*)d_in[10];
  const float* Wo = (const float*)d_in[11];
  const float* w0 = (const float*)d_in[12];
  const float* w1 = (const float*)d_in[13];
  const float* w2 = (const float*)d_in[14];
  const float* a0 = (const float*)d_in[15];
  const float* a1 = (const float*)d_in[16];
  const float* a2 = (const float*)d_in[17];
  const float* v0 = (const float*)d_in[18];
  const float* v1 = (const float*)d_in[19];
  const float* v2 = (const float*)d_in[20];
  const float* g1 = (const float*)d_in[21];
  const float* g2 = (const float*)d_in[22];
  const float* k_k = (const float*)d_in[23];
  const float* k_a = (const float*)d_in[24];
  const float* r_k = (const float*)d_in[25];
  const float* gn_w = (const float*)d_in[26];
  const float* gn_b = (const float*)d_in[27];

  char* ws = (char*)d_ws;
  size_t off = 0;
  auto alloc = [&](size_t bytes) -> void* {
    void* p = ws + off;
    off += (bytes + 255) & ~(size_t)255;
    return p;
  };

  const size_t BTC = (size_t)MM * CC;
  // bf16 activations
  unsigned short* xr_b = (unsigned short*)alloc(BTC * 2);
  unsigned short* xw_b = (unsigned short*)alloc(BTC * 2);
  unsigned short* xk_b = (unsigned short*)alloc(BTC * 2);
  unsigned short* xv_b = (unsigned short*)alloc(BTC * 2);
  unsigned short* xa_b = (unsigned short*)alloc(BTC * 2);
  unsigned short* xg_b = (unsigned short*)alloc(BTC * 2);
  // bf16 weights ([N,K] layout)
  unsigned short* Wr_b = (unsigned short*)alloc((size_t)CC * CC * 2);
  unsigned short* Wk_b = (unsigned short*)alloc((size_t)CC * CC * 2);
  unsigned short* Wv_b = (unsigned short*)alloc((size_t)CC * CC * 2);
  unsigned short* Wo_b = (unsigned short*)alloc((size_t)CC * CC * 2);
  unsigned short* w1t = (unsigned short*)alloc((size_t)64 * CC * 2);
  unsigned short* w2t = (unsigned short*)alloc((size_t)CC * 64 * 2);
  unsigned short* a1t = (unsigned short*)alloc((size_t)64 * CC * 2);
  unsigned short* a2t = (unsigned short*)alloc((size_t)CC * 64 * 2);
  unsigned short* v1t = (unsigned short*)alloc((size_t)32 * CC * 2);
  unsigned short* v2t = (unsigned short*)alloc((size_t)CC * 32 * 2);
  unsigned short* g1t = (unsigned short*)alloc((size_t)128 * CC * 2);
  unsigned short* g2t = (unsigned short*)alloc((size_t)CC * 128 * 2);
  // f32 intermediates
  float* r_f    = (float*)alloc(BTC * 4);
  float* k_f    = (float*)alloc(BTC * 4);   // k, then k2 in place
  float* v_f    = (float*)alloc(BTC * 4);   // v_raw, then mixed v in place
  float* wlog_f = (float*)alloc(BTC * 4);
  float* a_f    = (float*)alloc(BTC * 4);
  float* vmix_f = (float*)alloc(BTC * 4);   // sigmoid(v0+..), reused as o after prep
  float* g_f    = (float*)alloc(BTC * 4);
  float* nkk_f  = (float*)alloc(BTC * 4);
  float* kkah_f = (float*)alloc(BTC * 4);
  // small GEMM intermediates (bf16)
  unsigned short* t1_b = (unsigned short*)alloc((size_t)MM * 64 * 2);
  unsigned short* ta_b = (unsigned short*)alloc((size_t)MM * 64 * 2);
  unsigned short* tv_b = (unsigned short*)alloc((size_t)MM * 32 * 2);
  unsigned short* tg_b = (unsigned short*)alloc((size_t)MM * 128 * 2);
  unsigned short* y_b  = (unsigned short*)alloc(BTC * 2);
  float* o_f = vmix_f;  // reuse after prep consumed vmix

  const int EB = 256;
  const int nElem = (int)BTC;

  // 1) token shift + mixes
  mix_kernel<<<(nElem + EB - 1) / EB, EB, 0, stream>>>(
      x, x_r, x_w, x_k, x_v, x_a, x_g, xr_b, xw_b, xk_b, xv_b, xa_b, xg_b);

  // 2) weight conversions
  cvt_bf16<<<(CC * CC + EB - 1) / EB, EB, 0, stream>>>(Wr, Wr_b, CC * CC);
  cvt_bf16<<<(CC * CC + EB - 1) / EB, EB, 0, stream>>>(Wk, Wk_b, CC * CC);
  cvt_bf16<<<(CC * CC + EB - 1) / EB, EB, 0, stream>>>(Wv, Wv_b, CC * CC);
  cvt_bf16<<<(CC * CC + EB - 1) / EB, EB, 0, stream>>>(Wo, Wo_b, CC * CC);
  cvt_bf16_tr<<<(CC * 64 + EB - 1) / EB, EB, 0, stream>>>(w1, w1t, CC, 64);
  cvt_bf16_tr<<<(CC * 64 + EB - 1) / EB, EB, 0, stream>>>(w2, w2t, 64, CC);
  cvt_bf16_tr<<<(CC * 64 + EB - 1) / EB, EB, 0, stream>>>(a1, a1t, CC, 64);
  cvt_bf16_tr<<<(CC * 64 + EB - 1) / EB, EB, 0, stream>>>(a2, a2t, 64, CC);
  cvt_bf16_tr<<<(CC * 32 + EB - 1) / EB, EB, 0, stream>>>(v1, v1t, CC, 32);
  cvt_bf16_tr<<<(CC * 32 + EB - 1) / EB, EB, 0, stream>>>(v2, v2t, 32, CC);
  cvt_bf16_tr<<<(CC * 128 + EB - 1) / EB, EB, 0, stream>>>(g1, g1t, CC, 128);
  cvt_bf16_tr<<<(CC * 128 + EB - 1) / EB, EB, 0, stream>>>(g2, g2t, 128, CC);

  // 3) WMMA GEMMs. Big/wide GEMMs: 64x64 wave tile; narrow low-rank: 32x32.
  const dim3 gBig(MM / 64, CC / 64);      // 64 x 16
  wmma_gemm_bf16<4, 4><<<gBig, 32, 0, stream>>>(xr_b, Wr_b, r_f, MM, CC, CC, 0, nullptr, 1.f, 0);
  wmma_gemm_bf16<4, 4><<<gBig, 32, 0, stream>>>(xk_b, Wk_b, k_f, MM, CC, CC, 0, nullptr, 1.f, 0);
  wmma_gemm_bf16<4, 4><<<gBig, 32, 0, stream>>>(xv_b, Wv_b, v_f, MM, CC, CC, 0, nullptr, 1.f, 0);

  // decay path: t1 = tanh(xw@w1) [bf16], wlog = LOG_DECAY*sigmoid(w0 + t1@w2)
  wmma_gemm_bf16<2, 2><<<dim3(MM / 32, 64 / 32), 32, 0, stream>>>(
      xw_b, w1t, t1_b, MM, 64, CC, 1, nullptr, 1.f, 1);
  wmma_gemm_bf16<4, 4><<<gBig, 32, 0, stream>>>(
      t1_b, w2t, wlog_f, MM, CC, 64, 2, w0, LOG_DECAY_SCALE, 0);

  // a path: sigmoid(a0 + (xa@a1)@a2)
  wmma_gemm_bf16<2, 2><<<dim3(MM / 32, 64 / 32), 32, 0, stream>>>(
      xa_b, a1t, ta_b, MM, 64, CC, 0, nullptr, 1.f, 1);
  wmma_gemm_bf16<4, 4><<<gBig, 32, 0, stream>>>(ta_b, a2t, a_f, MM, CC, 64, 2, a0, 1.f, 0);

  // v-mix path: sigmoid(v0 + (xv@v1)@v2)
  wmma_gemm_bf16<2, 2><<<dim3(MM / 32, 32 / 32), 32, 0, stream>>>(
      xv_b, v1t, tv_b, MM, 32, CC, 0, nullptr, 1.f, 1);
  wmma_gemm_bf16<4, 4><<<gBig, 32, 0, stream>>>(tv_b, v2t, vmix_f, MM, CC, 32, 2, v0, 1.f, 0);

  // gate path: g = sigmoid(xg@g1)@g2
  wmma_gemm_bf16<2, 2><<<dim3(MM / 32, 128 / 32), 32, 0, stream>>>(
      xg_b, g1t, tg_b, MM, 128, CC, 2, nullptr, 1.f, 1);
  wmma_gemm_bf16<4, 4><<<gBig, 32, 0, stream>>>(tg_b, g2t, g_f, MM, CC, 128, 0, nullptr, 1.f, 0);

  // 4) prep: kk norm, -kk, kk*a, k2 (in-place), value mixing (in-place)
  prep_kernel<<<(MM * HH) / 4, 256, 0, stream>>>(
      k_f, a_f, v_f, v_first, vmix_f, nkk_f, kkah_f, k_k, k_a);

  // 5) sequential scan (state in registers)
  scan_kernel<<<BB * HH, DD, 0, stream>>>(r_f, wlog_f, k_f, v_f, nkk_f, kkah_f, o_f);

  // 6) groupnorm + bonus + gate -> bf16 y
  norm_bonus_gate<<<(MM * HH) / 4, 256, 0, stream>>>(
      o_f, r_f, k_f, v_f, g_f, r_k, gn_w, gn_b, y_b);

  // 7) out = y @ Wo^T  (f32 into first half of d_out)
  wmma_gemm_bf16<4, 4><<<gBig, 32, 0, stream>>>(y_b, Wo_b, d_out, MM, CC, CC, 0, nullptr, 1.f, 0);

  // 8) second tuple element: passthrough v_first
  (void)hipMemcpyAsync((float*)d_out + BTC, v_first, BTC * sizeof(float),
                       hipMemcpyDeviceToDevice, stream);
}
